// EncoderBlock_79688823210790
// MI455X (gfx1250) — matrix operand order repaired and checked by reading, not compiled, not run on hardware
//
#include <hip/hip_runtime.h>
#include <stdint.h>

// ---------------------------------------------------------------------------
// Types for CDNA5 WMMA (wave32)
// ---------------------------------------------------------------------------
typedef __attribute__((ext_vector_type(16))) __bf16 v16bf;
typedef __attribute__((ext_vector_type(8)))  float  v8f;
typedef __attribute__((ext_vector_type(4)))  unsigned int u32x4;

union FragBF { u32x4 q[2]; v16bf v; };

__device__ __forceinline__ unsigned short f2bf_raw(float f) {
  union { float f; unsigned int u; } c; c.f = f;
  unsigned int u = c.u;
  unsigned int r = (u + 0x7FFFu + ((u >> 16) & 1u)) >> 16;   // round-nearest-even
  return (unsigned short)r;
}

// ---------------------------------------------------------------------------
// fp32 -> bf16 conversion (plain)
// ---------------------------------------------------------------------------
__global__ __launch_bounds__(256)
void cvt_f32_to_bf16(const float* __restrict__ in, unsigned short* __restrict__ out,
                     long long n) {
  long long i = (long long)blockIdx.x * 256 + threadIdx.x;
  if (i < n) out[i] = f2bf_raw(in[i]);
}

// ---------------------------------------------------------------------------
// fp32 [R,C] -> bf16 transposed [C,R], LDS-tiled 32x32 (both sides coalesced).
// grid: (C/32, R/32, batch), block: 256.
// ---------------------------------------------------------------------------
__global__ __launch_bounds__(256)
void cvt_transpose_bf16(const float* __restrict__ in, unsigned short* __restrict__ out,
                        int R, int C, long long sIn, long long sOut) {
  __shared__ float t[32][33];
  const float* ip = in + (long long)blockIdx.z * sIn;
  unsigned short* op = out + (long long)blockIdx.z * sOut;
  const int lx = threadIdx.x & 31, ly = threadIdx.x >> 5;   // 32 x 8
  #pragma unroll
  for (int j = 0; j < 4; ++j) {
    int r = blockIdx.y * 32 + ly + j * 8;
    t[ly + j * 8][lx] = ip[(long long)r * C + blockIdx.x * 32 + lx];
  }
  __syncthreads();
  #pragma unroll
  for (int j = 0; j < 4; ++j) {
    int c = blockIdx.x * 32 + ly + j * 8;
    op[(long long)c * R + blockIdx.y * 32 + lx] = f2bf_raw(t[lx][ly + j * 8]);
  }
}

// ---------------------------------------------------------------------------
// GEMM:  C[M,N] = act( scale * (A_bf16[M,K] @ B_bf16[N,K]^T) + bias )
//   B is ALWAYS [N,K] row-major (weights pre-transposed; K-matrix natural).
//   OUTMODE: 0 = fp32 [M,N], 1 = bf16 [M,N], 2 = bf16 transposed [N,M]
//            (mode 2 packs each C-fragment column into one b128 store).
// Block: 256 thr (8 waves), tile 128(M) x 128(N), K-step 32, LDS double-buffer.
// Wave grid 4x2; each wave owns 32x64 = 2x4 WMMA 16x16 tiles (8 wmma/K-step).
// ---------------------------------------------------------------------------
template <int OUTMODE, int RELU>
__global__ __launch_bounds__(256)
void gemm_wmma_nt(const unsigned short* __restrict__ A,
                  const unsigned short* __restrict__ Bm,
                  float* __restrict__ Cf,
                  unsigned short* __restrict__ Cb,
                  const float* __restrict__ bias,
                  int K, int lda, int ldb, int ldc,
                  long long sA, long long sB, long long sC,
                  float scale)
{
  // row stride 40 halfwords (80B): 16B-aligned b128 ops, bank-spread
  __shared__ __align__(16) unsigned short As[2][128 * 40];
  __shared__ __align__(16) unsigned short Bt[2][128 * 40];

  const long long bz = blockIdx.z;
  const unsigned short* Ab = A  + bz * sA;
  const unsigned short* Bb = Bm + bz * sB;

  const int m0   = blockIdx.y * 128;
  const int n0   = blockIdx.x * 128;
  const int tid  = threadIdx.x;
  const int lane = tid & 31;
  const int wid  = tid >> 5;
  const int waveM = wid >> 1;        // 0..3 -> 32 rows each
  const int waveN = wid & 1;         // 0..1 -> 64 cols each
  const int hlf  = lane >> 4;        // ISA fragment lane-half
  const int lr   = lane & 15;

  v8f acc[2][4] = {};

  // staging: each thread copies 16 elems of A-tile and 16 of B-tile (2x b128 each)
  const int srow = tid >> 1;         // 0..127
  const int sseg = (tid & 1) << 4;   // 0 or 16 halfwords
  const unsigned short* ap = Ab + (long long)(m0 + srow) * lda + sseg;
  const unsigned short* bp = Bb + (long long)(n0 + srow) * ldb + sseg;

  // prologue: stage tile 0 into buffer 0
  u32x4 ra0 = *(const u32x4*)ap;
  u32x4 ra1 = *(const u32x4*)(ap + 8);
  u32x4 rb0 = *(const u32x4*)bp;
  u32x4 rb1 = *(const u32x4*)(bp + 8);
  *(u32x4*)&As[0][srow * 40 + sseg]     = ra0;
  *(u32x4*)&As[0][srow * 40 + sseg + 8] = ra1;
  *(u32x4*)&Bt[0][srow * 40 + sseg]     = rb0;
  *(u32x4*)&Bt[0][srow * 40 + sseg + 8] = rb1;
  __syncthreads();

  const int nk = K >> 5;
  for (int i = 0; i < nk; ++i) {
    const int cur = i & 1;
    const bool more = (i + 1 < nk);
    if (more) {            // issue next tile's global loads early (latency hiding)
      ap += 32; bp += 32;
      ra0 = *(const u32x4*)ap;  ra1 = *(const u32x4*)(ap + 8);
      rb0 = *(const u32x4*)bp;  rb1 = *(const u32x4*)(bp + 8);
      __builtin_prefetch((const void*)(ap + 32), 0, 1);
      __builtin_prefetch((const void*)(bp + 32), 0, 1);
    }

    // fragments per ISA 16-bit A/B layouts
    FragBF fa[2], fb[4];
    #pragma unroll
    for (int tm = 0; tm < 2; ++tm) {
      const int r = waveM * 32 + tm * 16 + lr;               // A row M = lane%16
      fa[tm].q[0] = *(const u32x4*)&As[cur][r * 40 + hlf * 8];
      fa[tm].q[1] = *(const u32x4*)&As[cur][r * 40 + 16 + hlf * 8];
    }
    #pragma unroll
    for (int tn = 0; tn < 4; ++tn) {
      const int r = waveN * 64 + tn * 16 + lr;               // B col N = lane%16
      fb[tn].q[0] = *(const u32x4*)&Bt[cur][r * 40 + hlf * 8];
      fb[tn].q[1] = *(const u32x4*)&Bt[cur][r * 40 + 16 + hlf * 8];
    }

    #pragma unroll
    for (int tm = 0; tm < 2; ++tm)
      #pragma unroll
      for (int tn = 0; tn < 4; ++tn)
        acc[tm][tn] = __builtin_amdgcn_wmma_f32_16x16x32_bf16(
            false, fa[tm].v, false, fb[tn].v, (short)0, acc[tm][tn], false, false);

    if (more) {            // write-back next tile into the other buffer
      const int nxt = cur ^ 1;
      *(u32x4*)&As[nxt][srow * 40 + sseg]     = ra0;
      *(u32x4*)&As[nxt][srow * 40 + sseg + 8] = ra1;
      *(u32x4*)&Bt[nxt][srow * 40 + sseg]     = rb0;
      *(u32x4*)&Bt[nxt][srow * 40 + sseg + 8] = rb1;
    }
    __syncthreads();
  }

  // ---- epilogue: C frag: VGPR r -> row (lane/16)*8 + r, col = lane%16
  #pragma unroll
  for (int tm = 0; tm < 2; ++tm) {
    #pragma unroll
    for (int tn = 0; tn < 4; ++tn) {
      const int col  = n0 + waveN * 64 + tn * 16 + lr;
      const int rowb = m0 + waveM * 32 + tm * 16 + hlf * 8;
      const float bv = bias ? bias[col] : 0.0f;
      if (OUTMODE == 2) {
        // transposed bf16: 8 contiguous elems along M -> single b128 store
        union { u32x4 q; unsigned short s[8]; } pk;
        #pragma unroll
        for (int r = 0; r < 8; ++r) {
          float v = acc[tm][tn][r] * scale + bv;
          if (RELU) v = fmaxf(v, 0.0f);
          pk.s[r] = f2bf_raw(v);
        }
        *(u32x4*)&Cb[bz * sC + (long long)col * ldc + rowb] = pk.q;
      } else {
        #pragma unroll
        for (int r = 0; r < 8; ++r) {
          float v = acc[tm][tn][r] * scale + bv;
          if (RELU) v = fmaxf(v, 0.0f);
          const long long idx = bz * sC + (long long)(rowb + r) * ldc + col;
          if (OUTMODE == 0) Cf[idx] = v;
          else              Cb[idx] = f2bf_raw(v);
        }
      }
    }
  }
}

// ---------------------------------------------------------------------------
// Row softmax over fp32 [rows, 2048] -> bf16 probabilities
// ---------------------------------------------------------------------------
__global__ __launch_bounds__(256)
void softmax_bf16(const float* __restrict__ Sf, unsigned short* __restrict__ P) {
  const int n = 2048;
  const long long row = blockIdx.x;
  const float* x = Sf + row * (long long)n;
  unsigned short* y = P + row * (long long)n;
  const int tid = threadIdx.x, lane = tid & 31, wid = tid >> 5;

  float v[8];
  float mx = -3.4e38f;
  #pragma unroll
  for (int j = 0; j < 8; ++j) { v[j] = x[j * 256 + tid]; mx = fmaxf(mx, v[j]); }
  #pragma unroll
  for (int o = 16; o > 0; o >>= 1) mx = fmaxf(mx, __shfl_xor(mx, o, 32));
  __shared__ float r1[8];
  if (lane == 0) r1[wid] = mx;
  __syncthreads();
  float m2 = r1[0];
  #pragma unroll
  for (int w = 1; w < 8; ++w) m2 = fmaxf(m2, r1[w]);

  float sum = 0.0f;
  #pragma unroll
  for (int j = 0; j < 8; ++j) { v[j] = __expf(v[j] - m2); sum += v[j]; }
  #pragma unroll
  for (int o = 16; o > 0; o >>= 1) sum += __shfl_xor(sum, o, 32);
  __shared__ float r2[8];
  if (lane == 0) r2[wid] = sum;
  __syncthreads();
  float tot = 0.0f;
  #pragma unroll
  for (int w = 0; w < 8; ++w) tot += r2[w];
  const float inv = 1.0f / tot;
  #pragma unroll
  for (int j = 0; j < 8; ++j) y[j * 256 + tid] = f2bf_raw(v[j] * inv);
}

// ---------------------------------------------------------------------------
// out = LayerNorm(A + B) * g + be over rows of length 1024.
// ---------------------------------------------------------------------------
__global__ __launch_bounds__(256)
void add_ln_kernel(const float* __restrict__ A, const float* __restrict__ Bv,
                   const float* __restrict__ g, const float* __restrict__ be,
                   float* __restrict__ outf, unsigned short* __restrict__ outb)
{
  const int n = 1024;
  const long long row = blockIdx.x;
  const float* a = A  + row * (long long)n;
  const float* b = Bv + row * (long long)n;
  const int tid = threadIdx.x, lane = tid & 31, wid = tid >> 5;

  float v[4]; float s = 0.0f;
  #pragma unroll
  for (int j = 0; j < 4; ++j) { int i = j * 256 + tid; v[j] = a[i] + b[i]; s += v[j]; }
  #pragma unroll
  for (int o = 16; o > 0; o >>= 1) s += __shfl_xor(s, o, 32);
  __shared__ float r1[8];
  if (lane == 0) r1[wid] = s;
  __syncthreads();
  float tot = 0.0f;
  #pragma unroll
  for (int w = 0; w < 8; ++w) tot += r1[w];
  const float mean = tot * (1.0f / 1024.0f);

  float ss = 0.0f;
  #pragma unroll
  for (int j = 0; j < 4; ++j) { float d = v[j] - mean; ss += d * d; }
  #pragma unroll
  for (int o = 16; o > 0; o >>= 1) ss += __shfl_xor(ss, o, 32);
  __shared__ float r2[8];
  if (lane == 0) r2[wid] = ss;
  __syncthreads();
  float tv = 0.0f;
  #pragma unroll
  for (int w = 0; w < 8; ++w) tv += r2[w];
  const float rstd = rsqrtf(tv * (1.0f / 1024.0f) + 1e-5f);

  #pragma unroll
  for (int j = 0; j < 4; ++j) {
    int i = j * 256 + tid;
    float o = (v[j] - mean) * rstd * g[i] + be[i];
    outf[row * (long long)n + i] = o;
    if (outb) outb[row * (long long)n + i] = f2bf_raw(o);
  }
}

// ---------------------------------------------------------------------------
// Host-side orchestration
// ---------------------------------------------------------------------------
extern "C" void kernel_launch(void* const* d_in, const int* in_sizes, int n_in,
                              void* d_out, int out_size, void* d_ws, size_t ws_size,
                              hipStream_t stream) {
  const float* x   = (const float*)d_in[0];
  const float* Wq  = (const float*)d_in[1];
  const float* Wk  = (const float*)d_in[2];
  const float* Wv  = (const float*)d_in[3];
  const float* Wo  = (const float*)d_in[4];
  const float* bo  = (const float*)d_in[5];
  const float* w1  = (const float*)d_in[6];
  const float* b1  = (const float*)d_in[7];
  const float* w2  = (const float*)d_in[8];
  const float* b2  = (const float*)d_in[9];
  const float* w3  = (const float*)d_in[10];
  const float* b3  = (const float*)d_in[11];
  const float* w4  = (const float*)d_in[12];
  const float* b4  = (const float*)d_in[13];
  const float* g1  = (const float*)d_in[14];
  const float* be1 = (const float*)d_in[15];
  const float* g2  = (const float*)d_in[16];
  const float* be2 = (const float*)d_in[17];
  float* out = (float*)d_out;

  const long long Bn = 4, S = 2048, D = 1024, H = 8, HID = 4096;
  const long long M = Bn * S;                 // 8192 token rows

  // ---- bump allocator over d_ws
  char* ws = (char*)d_ws;
  size_t off = 0;
  auto alloc = [&](size_t bytes) -> void* {
    off = (off + 255) & ~(size_t)255;
    void* p = ws + off; off += bytes; return p;
  };
  unsigned short* Xb  = (unsigned short*)alloc(M * D * 2);
  unsigned short* WqT = (unsigned short*)alloc(H * D * D * 2);   // [h][e_out][e_in]
  unsigned short* WkT = (unsigned short*)alloc(H * D * D * 2);
  unsigned short* WvT = (unsigned short*)alloc(H * D * D * 2);
  unsigned short* WoT = (unsigned short*)alloc(H * D * D * 2);   // [D][H*D]
  unsigned short* w1T = (unsigned short*)alloc(D * HID * 2);     // [HID][D]
  unsigned short* w2T = (unsigned short*)alloc(HID * HID * 2);
  unsigned short* w3T = (unsigned short*)alloc(HID * HID * 2);
  unsigned short* w4T = (unsigned short*)alloc(HID * D * 2);     // [D][HID]
  unsigned short* Qb  = (unsigned short*)alloc(M * D * 2);       // per-head, reused
  unsigned short* Kb  = (unsigned short*)alloc(M * D * 2);
  unsigned short* Vt  = (unsigned short*)alloc(M * D * 2);       // transposed: [e][m]
  float*          Sf  = (float*)alloc(Bn * S * S * 4);
  unsigned short* Pb  = (unsigned short*)alloc(Bn * S * S * 2);
  unsigned short* Zb  = (unsigned short*)alloc(M * H * D * 2);
  float*          AOf = (float*)alloc(M * D * 4);
  float*          z1f = (float*)alloc(M * D * 4);
  unsigned short* z1b = (unsigned short*)alloc(M * D * 2);
  unsigned short* y1b = (unsigned short*)alloc(M * HID * 2);
  unsigned short* y2b = (unsigned short*)alloc(M * HID * 2);
  unsigned short* y3b = (unsigned short*)alloc(M * HID * 2);
  float*          y4f = (float*)alloc(M * D * 4);

  auto cvt = [&](const float* in, unsigned short* o, long long n) {
    unsigned nb = (unsigned)((n + 255) / 256);
    cvt_f32_to_bf16<<<dim3(nb), dim3(256), 0, stream>>>(in, o, n);
  };
  auto cvtT = [&](const float* in, unsigned short* o, int R, int C, int nb,
                  long long sIn, long long sOut) {
    dim3 grid((unsigned)(C / 32), (unsigned)(R / 32), (unsigned)nb);
    cvt_transpose_bf16<<<grid, dim3(256), 0, stream>>>(in, o, R, C, sIn, sOut);
  };
  auto gemm = [&](int outmode, int relu,
                  const unsigned short* A, const unsigned short* Bm,
                  void* C, const float* bias, int Mm, int Nn, int Kk,
                  int lda, int ldb, int ldc,
                  long long sA, long long sB, long long sC, int nb, float scale) {
    dim3 grid((unsigned)(Nn / 128), (unsigned)(Mm / 128), (unsigned)nb);
    if (outmode == 0)
      gemm_wmma_nt<0, 0><<<grid, dim3(256), 0, stream>>>(
          A, Bm, (float*)C, nullptr, bias, Kk, lda, ldb, ldc, sA, sB, sC, scale);
    else if (outmode == 1 && relu)
      gemm_wmma_nt<1, 1><<<grid, dim3(256), 0, stream>>>(
          A, Bm, nullptr, (unsigned short*)C, bias, Kk, lda, ldb, ldc, sA, sB, sC, scale);
    else if (outmode == 1)
      gemm_wmma_nt<1, 0><<<grid, dim3(256), 0, stream>>>(
          A, Bm, nullptr, (unsigned short*)C, bias, Kk, lda, ldb, ldc, sA, sB, sC, scale);
    else
      gemm_wmma_nt<2, 0><<<grid, dim3(256), 0, stream>>>(
          A, Bm, nullptr, (unsigned short*)C, bias, Kk, lda, ldb, ldc, sA, sB, sC, scale);
  };

  // ---- precision downcast (+ weight pre-transpose so every GEMM is A.B^T)
  cvt(x, Xb, M * D);
  cvtT(Wq, WqT, (int)D, (int)D, (int)H, D * D, D * D);
  cvtT(Wk, WkT, (int)D, (int)D, (int)H, D * D, D * D);
  cvtT(Wv, WvT, (int)D, (int)D, (int)H, D * D, D * D);
  cvtT(Wo, WoT, (int)(H * D), (int)D, 1, 0, 0);
  cvtT(w1, w1T, (int)D, (int)HID, 1, 0, 0);
  cvtT(w2, w2T, (int)HID, (int)HID, 1, 0, 0);
  cvtT(w3, w3T, (int)HID, (int)HID, 1, 0, 0);
  cvtT(w4, w4T, (int)HID, (int)D, 1, 0, 0);

  const float qscale = 1.0f / 32.0f;   // 1/sqrt(D), folded into Q

  // ---- attention, head by head (score/prob buffers reused across heads)
  for (int h = 0; h < 8; ++h) {
    const long long wOff = (long long)h * D * D;
    // Q/K normal bf16; V written TRANSPOSED so P@V consumes it as B=[N,K]
    gemm(1, 0, Xb, WqT + wOff, Qb, nullptr, 8192, 1024, 1024,
         1024, 1024, 1024, 0, 0, 0, 1, qscale);
    gemm(1, 0, Xb, WkT + wOff, Kb, nullptr, 8192, 1024, 1024,
         1024, 1024, 1024, 0, 0, 0, 1, 1.0f);
    gemm(2, 0, Xb, WvT + wOff, Vt, nullptr, 8192, 1024, 1024,
         1024, 1024, (int)M, 0, 0, 0, 1, 1.0f);          // Vt[e][m], ldct = M
    // scores[b] = Q[b] @ K[b]^T  (K natural [t][e] == [N,K])
    gemm(0, 0, Qb, Kb, Sf, nullptr, 2048, 2048, 1024,
         1024, 1024, 2048, S * D, S * D, S * S, (int)Bn, 1.0f);
    softmax_bf16<<<dim3((unsigned)(Bn * S)), dim3(256), 0, stream>>>(Sf, Pb);
    // Z[b,:,h*D:(h+1)*D] = P[b] @ V[b]   (B = Vt[e][b*S+t], [N,K] slice)
    gemm(1, 0, Pb, Vt, Zb + (long long)h * D, nullptr, 2048, 1024, 2048,
         2048, (int)M, (int)(H * D), S * S, S, S * H * D, (int)Bn, 1.0f);
  }

  // ---- output projection + residual + LN1
  gemm(0, 0, Zb, WoT, AOf, bo, 8192, 1024, 8192,
       (int)(H * D), (int)(H * D), 1024, 0, 0, 0, 1, 1.0f);
  add_ln_kernel<<<dim3((unsigned)M), dim3(256), 0, stream>>>(x, AOf, g1, be1, z1f, z1b);

  // ---- FFN
  gemm(1, 1, z1b, w1T, y1b, b1, 8192, 4096, 1024,
       1024, 1024, 4096, 0, 0, 0, 1, 1.0f);
  gemm(1, 1, y1b, w2T, y2b, b2, 8192, 4096, 4096,
       4096, 4096, 4096, 0, 0, 0, 1, 1.0f);
  gemm(1, 1, y2b, w3T, y3b, b3, 8192, 4096, 4096,
       4096, 4096, 4096, 0, 0, 0, 1, 1.0f);
  gemm(0, 0, y3b, w4T, y4f, b4, 8192, 1024, 4096,
       4096, 4096, 1024, 0, 0, 0, 1, 1.0f);

  // ---- residual + LN2 -> final output
  add_ln_kernel<<<dim3((unsigned)M), dim3(256), 0, stream>>>(z1f, y4f, g2, be2, out, nullptr);
}
